// ModelLossSemsegGatedCRF_16097537426201
// MI455X (gfx1250) — matrix operand order repaired and checked by reading, not compiled, not run on hardware
//
#include <hip/hip_runtime.h>
#include <hip/hip_bf16.h>

// ---------------------------------------------------------------------------
// Gated-CRF loss for y_hat_softmax [2,21,128,128] f32, rgb [2,3,256,256] f32.
// Output: single f32 scalar.
//
// loss = ( sum_w  -  sum_{p,t} w(p,t) * S(p,t) ) / (N*H*W)
//   S(p,t) = sum_c y[p,c]*y[p+t,c]   (zero-padded unfold)
//   w_in(p,t) = 0.9*exp(-0.5*(dxy2+drgb2)) + 0.1*exp(-0.5*dxy2)
//       dxy2 = (dx^2+dy^2)/36,  drgb2 = 100*||rgb_p-rgb_q||^2
//   OOB taps:  w_oob(p) = exp(-0.5*((x/6)^2+(y/6)^2)) *
//                         (0.9*exp(-50*||rgb_p||^2) + 0.1)
//
// S is computed with v_wmma_f32_16x16x32_bf16 as a banded correlation:
// A = 16 center pixels x 32 ch (bf16), B = 32 shifted pixels (2 tiles of 16).
// Y is pre-transposed once into pixel-major bf16 rows (21 ch + 11 zeros),
// padded 16 columns left/right with zeros, so WMMA fragments are direct
// global_load_b128s from the L2-resident workspace.
// ---------------------------------------------------------------------------

typedef __attribute__((ext_vector_type(16))) __bf16 v16bf;
typedef __attribute__((ext_vector_type(8)))  __bf16 v8bf;
typedef __attribute__((ext_vector_type(8)))  float  v8f;
typedef __attribute__((ext_vector_type(4)))  float  v4f;

#define CRF_N 2
#define CRF_C 21
#define CRF_H 128
#define CRF_W 128
#define CRF_R 5
#define XPAD  16
#define WPAD  (CRF_W + 2 * XPAD)      // 160 padded columns per row

// ---------------- pre-pass 1: 2x2 mean pool of rgb, times 10 ----------------
__global__ void crf_prep_rgb(const float* __restrict__ rgb, float* __restrict__ rgb10) {
  int idx = blockIdx.x * blockDim.x + threadIdx.x;          // n*H*W
  if (idx >= CRF_N * CRF_H * CRF_W) return;
  int x = idx & (CRF_W - 1);
  int y = (idx >> 7) & (CRF_H - 1);
  int n = idx >> 14;
  const float* base = rgb + (size_t)n * 3 * 256 * 256;
#pragma unroll
  for (int c = 0; c < 3; ++c) {
    const float* p = base + (size_t)c * 256 * 256;
    float s = p[(2 * y) * 256 + 2 * x] + p[(2 * y) * 256 + 2 * x + 1] +
              p[(2 * y + 1) * 256 + 2 * x] + p[(2 * y + 1) * 256 + 2 * x + 1];
    rgb10[idx * 4 + c] = 2.5f * s;    // mean * 10  (sigma_rgb = 0.1)
  }
  rgb10[idx * 4 + 3] = 0.0f;
}

// ---------------- pre-pass 2: transpose Y to padded pixel-major bf16 --------
// ypad layout: [n][y][xp=0..159][ch=0..31] bf16 (64B per pixel row).
__global__ void crf_prep_y(const float* __restrict__ ysm, __bf16* __restrict__ ypad) {
  int idx = blockIdx.x * blockDim.x + threadIdx.x;          // n*H*WPAD
  if (idx >= CRF_N * CRF_H * WPAD) return;
  int xp  = idx % WPAD;
  int rem = idx / WPAD;
  int y   = rem & (CRF_H - 1);
  int n   = rem >> 7;
  int x   = xp - XPAD;

  float row[32];
#pragma unroll
  for (int c = 0; c < 32; ++c) row[c] = 0.0f;
  if (x >= 0 && x < CRF_W) {
    const float* yb = ysm + ((size_t)n * CRF_C) * CRF_H * CRF_W + y * CRF_W + x;
#pragma unroll
    for (int c = 0; c < CRF_C; ++c) row[c] = yb[(size_t)c * CRF_H * CRF_W];
  }
  v16bf lo{}, hi{};
#pragma unroll
  for (int c = 0; c < 16; ++c) { lo[c] = (__bf16)row[c]; hi[c] = (__bf16)row[16 + c]; }
  __bf16* dst = ypad + (size_t)idx * 32;
  *(v16bf*)dst        = lo;
  *(v16bf*)(dst + 16) = hi;
}

// ---------------- main kernel: one block per (n, row), 8 waves x 16 px ------
__global__ __launch_bounds__(256) void crf_main(const __bf16* __restrict__ ypad,
                                                const float* __restrict__ rgb10,
                                                float* __restrict__ partials) {
  __shared__ float sred[8][2];

  const int nb   = blockIdx.x;            // 0..N*H-1
  const int n    = nb >> 7;
  const int yrow = nb & (CRF_H - 1);
  const int lane = threadIdx.x & 31;
  const int wv   = threadIdx.x >> 5;
  const int x0   = wv * 16;
  const int h    = lane >> 4;             // half: 0 or 1
  const int m    = lane & 15;
  const int j    = m;                     // D-tile column (neighbor index)

  const float* rbase = rgb10 + (size_t)(n * CRF_H + yrow) * CRF_W * 4;

  // A fragment (16-bit 16x32 layout): lane holds row M=m;
  //   elems 0..7 = K h*8..h*8+7, elems 8..15 = K 16+h*8..16+h*8+7
  const __bf16* arow = ypad + ((size_t)(n * CRF_H + yrow) * WPAD + XPAD + x0 + m) * 32;
  v8bf a_lo = *(const v8bf*)(arow + h * 8);
  v8bf a_hi = *(const v8bf*)(arow + 16 + h * 8);
  v16bf afrag = __builtin_shufflevector(a_lo, a_hi, 0, 1, 2, 3, 4, 5, 6, 7,
                                        8, 9, 10, 11, 12, 13, 14, 15);

  // center rgb (x10) for the 8 pixels this lane's D VGPRs correspond to
  float cr[8], cg[8], cb[8];
#pragma unroll
  for (int g = 0; g < 8; ++g) {
    int xi = x0 + g + 8 * h;
    cr[g] = rbase[xi * 4 + 0];
    cg[g] = rbase[xi * 4 + 1];
    cb[g] = rbase[xi * 4 + 2];
  }

  float acc_w = 0.0f, acc_ws = 0.0f;

  // OOB-tap closed form: n_oob(p)*w_oob(p) (lanes 0..15 -> one pixel each),
  // branchless: noob==0 for interior pixels.
  if (lane < 16) {
    int x  = x0 + m;
    int nx = min(x, CRF_R) + min(CRF_W - 1 - x, CRF_R) + 1;
    int ny = min(yrow, CRF_R) + min(CRF_H - 1 - yrow, CRF_R) + 1;
    float noob = (float)(121 - nx * ny);
    float fx = (float)x * (1.0f / 6.0f);
    float fy = (float)yrow * (1.0f / 6.0f);
    float r = rbase[x * 4 + 0], g2 = rbase[x * 4 + 1], b = rbase[x * 4 + 2];
    float woob = __expf(-0.5f * (fx * fx + fy * fy)) *
                 (0.9f * __expf(-0.5f * (r * r + g2 * g2 + b * b)) + 0.1f);
    acc_w += noob * woob;
  }

  for (int dy = -CRF_R; dy <= CRF_R; ++dy) {      // uniform across the block
    int yy = yrow + dy;
    if (yy < 0 || yy >= CRF_H) continue;          // uniform

    const __bf16* brow = ypad + ((size_t)(n * CRF_H + yy) * WPAD + XPAD) * 32;
    const v4f*   rnb4  = (const v4f*)(rgb10 + (size_t)(n * CRF_H + yy) * CRF_W * 4);

#pragma unroll
    for (int tile = 0; tile < 2; ++tile) {
      int xnb = x0 + tile * 16 + j - 5;           // neighbor x for this column
      // B fragment (32x16, K x N): lane holds column n=j; K = h*8..+7, 16+h*8..+7.
      // Padded-zero columns make OOB dots exactly 0.
      const __bf16* bp = brow + (size_t)xnb * 32;
      v8bf b_lo = *(const v8bf*)(bp + h * 8);
      v8bf b_hi = *(const v8bf*)(bp + 16 + h * 8);
      v16bf bfrag = __builtin_shufflevector(b_lo, b_hi, 0, 1, 2, 3, 4, 5, 6, 7,
                                            8, 9, 10, 11, 12, 13, 14, 15);
      v8f cacc = {};
      cacc = __builtin_amdgcn_wmma_f32_16x16x32_bf16(
          /*neg_a=*/false, afrag, /*neg_b=*/false, bfrag,
          /*c_mod=*/(short)0, cacc, /*reuse_a=*/false, /*reuse_b=*/false);

      bool colin = ((unsigned)xnb < (unsigned)CRF_W);
      int  xc    = min(max(xnb, 0), CRF_W - 1);   // clamped load, masked below
      v4f  nrgb  = rnb4[xc];

#pragma unroll
      for (int g = 0; g < 8; ++g) {
        int i  = g + 8 * h;                       // center pixel index in segment
        int dx = xnb - (x0 + i);
        bool band = ((unsigned)(dx + CRF_R) <= (unsigned)(2 * CRF_R));
        bool notc = !((dx == 0) & (dy == 0));
        float dr = cr[g] - nrgb.x, dg = cg[g] - nrgb.y, db = cb[g] - nrgb.z;
        float drgb = dr * dr + dg * dg + db * db;             // already x100
        float dxy2 = (float)(dx * dx + dy * dy) * (1.0f / 36.0f);
        float e1 = __expf(-0.5f * (dxy2 + drgb));
        float e2 = __expf(-0.5f * dxy2);
        float w  = 0.9f * e1 + 0.1f * e2;
        w = (colin & band & notc) ? w : 0.0f;     // branchless validity mask
        acc_w  += w;
        acc_ws += w * cacc[g];                    // w * S(p,t)
      }
    }
  }

  // deterministic reduction: wave shfl-xor -> LDS -> block partial
#pragma unroll
  for (int off = 16; off > 0; off >>= 1) {
    acc_w  += __shfl_xor(acc_w,  off, 32);
    acc_ws += __shfl_xor(acc_ws, off, 32);
  }
  if (lane == 0) { sred[wv][0] = acc_w; sred[wv][1] = acc_ws; }
  __syncthreads();
  if (threadIdx.x == 0) {
    float sw = 0.0f, sws = 0.0f;
    for (int w2 = 0; w2 < 8; ++w2) { sw += sred[w2][0]; sws += sred[w2][1]; }
    partials[blockIdx.x * 2 + 0] = sw;
    partials[blockIdx.x * 2 + 1] = sws;
  }
}

// ---------------- final fixed-order reduction -------------------------------
__global__ void crf_reduce(const float* __restrict__ partials, float* __restrict__ out) {
  if (threadIdx.x == 0 && blockIdx.x == 0) {
    float sw = 0.0f, sws = 0.0f;
    for (int i = 0; i < CRF_N * CRF_H; ++i) {
      sw  += partials[2 * i + 0];
      sws += partials[2 * i + 1];
    }
    out[0] = (sw - sws) / (float)(CRF_N * CRF_H * CRF_W);
  }
}

extern "C" void kernel_launch(void* const* d_in, const int* in_sizes, int n_in,
                              void* d_out, int out_size, void* d_ws, size_t ws_size,
                              hipStream_t stream) {
  (void)in_sizes; (void)n_in; (void)out_size; (void)ws_size;
  const float* ysm = (const float*)d_in[0];   // [2,21,128,128] f32
  const float* rgb = (const float*)d_in[1];   // [2,3,256,256]  f32

  // workspace layout:
  //   [rgb10  : N*H*W*4 floats              = 512 KB]
  //   [ypad   : N*H*WPAD*32 bf16            = 2.62 MB]
  //   [partials: N*H*2 floats               = 2 KB]
  size_t off_rgb10 = 0;
  size_t off_ypad  = off_rgb10 + (size_t)CRF_N * CRF_H * CRF_W * 4 * sizeof(float);
  size_t off_part  = off_ypad  + (size_t)CRF_N * CRF_H * WPAD * 32 * sizeof(__bf16);

  float*  rgb10    = (float*)((char*)d_ws + off_rgb10);
  __bf16* ypad     = (__bf16*)((char*)d_ws + off_ypad);
  float*  partials = (float*)((char*)d_ws + off_part);

  int npix = CRF_N * CRF_H * CRF_W;
  int npad = CRF_N * CRF_H * WPAD;
  crf_prep_rgb<<<(npix + 255) / 256, 256, 0, stream>>>(rgb, rgb10);
  crf_prep_y<<<(npad + 255) / 256, 256, 0, stream>>>(ysm, ypad);
  crf_main<<<CRF_N * CRF_H, 256, 0, stream>>>(ypad, rgb10, partials);
  crf_reduce<<<1, 32, 0, stream>>>(partials, (float*)d_out);
}
